// MultiHeadSelfAttention_30185030156731
// MI455X (gfx1250) — matrix operand order repaired and checked by reading, not compiled, not run on hardware
//
#include <hip/hip_runtime.h>
#include <hip/hip_bf16.h>

#define B_ 2
#define S_ 2048
#define D_ 1024
#define H_ 16
#define DK_ 64
#define NEGV -1000000000.0f

typedef __bf16 bf16;
typedef __attribute__((ext_vector_type(16))) __bf16 v16bf;
typedef __attribute__((ext_vector_type(8)))  float v8f;
typedef unsigned int u32x4 __attribute__((ext_vector_type(4)));
typedef int i32x4 __attribute__((ext_vector_type(4)));
typedef int i32x8 __attribute__((ext_vector_type(8)));

#if defined(__has_builtin)
#if __has_builtin(__builtin_amdgcn_tensor_load_to_lds) && \
    __has_builtin(__builtin_amdgcn_s_wait_tensorcnt)
#define USE_TDM 1
#endif
#endif
#ifndef USE_TDM
#define USE_TDM 0
#endif

static __device__ __forceinline__ v8f wmma_bf16(v16bf a, v16bf b, v8f c) {
  return __builtin_amdgcn_wmma_f32_16x16x32_bf16(false, a, false, b, (short)0, c,
                                                 false, false);
}

// ---- WMMA fragment loaders (layouts per CDNA5 ISA §7.12.2) ------
// A 16x32 bf16: lane%16 = row M; K pairs: v<4 -> 2v, v>=4 -> 2v+8; +8*(lane/16)
static __device__ __forceinline__ v16bf load_a_bf16(const bf16* __restrict__ src,
                                                    int ld, int m0, int k0, int lane) {
  const int r = lane & 15, hh = lane >> 4;
  const bf16* row = src + (size_t)(m0 + r) * ld + (k0 + hh * 8);
  v16bf a;
#pragma unroll
  for (int v = 0; v < 8; ++v) {
    const int kk = (v < 4) ? (2 * v) : (8 + 2 * v);  // 0,2,4,6,16,18,20,22
    a[2 * v]     = row[kk];
    a[2 * v + 1] = row[kk + 1];
  }
  return a;
}

// B 32x16 from a [N][K]-layout tile (LDS): lane%16 = column N;
// K pairs: v<4 -> 2v, v>=4 -> 2v+8; +8*(lane/16) (same as A).
static __device__ __forceinline__ v16bf load_b_lds(const bf16* tile, int ldk,
                                                   int n0, int koff, int lane) {
  const int c = lane & 15, hh = lane >> 4;
  const bf16* row = tile + (n0 + c) * ldk + koff + hh * 8;
  v16bf b;
#pragma unroll
  for (int v = 0; v < 8; ++v) {
    const int kk = (v < 4) ? (2 * v) : (8 + 2 * v);
    b[2 * v]     = row[kk];
    b[2 * v + 1] = row[kk + 1];
  }
  return b;
}

// ---- Tensor Data Mover: 2D bf16 tile -> LDS (D# per CDNA5 ISA) -
#if USE_TDM
static __device__ __forceinline__ void tdm_load_tile_bf16(
    const bf16* gsrc_tile, unsigned int lds_byte_off,
    unsigned int tile_cols, unsigned int tile_rows, unsigned int row_stride) {
  const unsigned long long ga = (unsigned long long)(uintptr_t)gsrc_tile;
  u32x4 g0;
  g0[0] = 1u;                                            // count=1 (user D#)
  g0[1] = lds_byte_off;                                  // lds_addr
  g0[2] = (unsigned int)ga;                              // global_addr[31:0]
  g0[3] = (unsigned int)((ga >> 32) & 0x01FFFFFFu)       // global_addr[56:32]
          | (2u << 30);                                  // type=2 ("image")
  i32x8 g1;
  g1[0] = (int)(1u << 16);                               // data_size=1 (2B), mask=0
  g1[1] = (int)((tile_cols & 0xFFFFu) << 16);            // tensor_dim0[15:0]
  g1[2] = (int)(((tile_cols >> 16) & 0xFFFFu) |
                ((tile_rows & 0xFFFFu) << 16));          // dim0[31:16] | dim1[15:0]
  g1[3] = (int)(((tile_rows >> 16) & 0xFFFFu) |
                ((tile_cols & 0xFFFFu) << 16));          // dim1[31:16] | tile_dim0
  g1[4] = (int)(tile_rows & 0xFFFFu);                    // tile_dim1; tile_dim2=0
  g1[5] = (int)row_stride;                               // tensor_dim0_stride[31:0]
  g1[6] = 0;
  g1[7] = 0;
  i32x4 z4 = {0, 0, 0, 0};
#if __clang_major__ >= 23
  i32x8 z8 = {0, 0, 0, 0, 0, 0, 0, 0};
  __builtin_amdgcn_tensor_load_to_lds(g0, g1, z4, z4, z8, 0);
#else
  __builtin_amdgcn_tensor_load_to_lds(g0, g1, z4, z4, 0);
#endif
}
#endif

static __device__ __forceinline__ void stage_issue(const bf16* __restrict__ g,
                                                   bf16* s, int tile_cols,
                                                   int tile_rows, int stride,
                                                   int wv) {
#if USE_TDM
  if (wv == 0)
    tdm_load_tile_bf16(g, (unsigned int)(uintptr_t)s, tile_cols, tile_rows, stride);
#else
  (void)wv;  // cooperative synchronous copy by all 256 threads
  const int total = tile_cols * tile_rows;
  for (int i = (int)threadIdx.x * 2; i < total; i += 256 * 2) {
    const int r = i / tile_cols, cc = i % tile_cols;
    *(unsigned int*)(s + i) = *(const unsigned int*)(g + (size_t)r * stride + cc);
  }
#endif
}

static __device__ __forceinline__ void stage_wait(int wv) {
#if USE_TDM
  if (wv == 0) __builtin_amdgcn_s_wait_tensorcnt(0);
#else
  (void)wv;
#endif
}

// ---- kernel 0: fp32 -> bf16 conversion ------------------------------------
__global__ __launch_bounds__(256) void cvt_bf16_kernel(const float* __restrict__ src,
                                                       bf16* __restrict__ dst, int n) {
  const int i = ((int)blockIdx.x * 256 + (int)threadIdx.x) * 4;
  if (i < n) {
    const float4 f = *(const float4*)(src + i);
    dst[i + 0] = (bf16)f.x;
    dst[i + 1] = (bf16)f.y;
    dst[i + 2] = (bf16)f.z;
    dst[i + 3] = (bf16)f.w;
  }
}

// ---- kernel 1: fused Q/K/V projections (grid.z = which) --------------------
// Block = 8 waves x (32 M-rows x 64 N-cols each); weight tile (64x32) TDM-staged
// in LDS, double buffered; 8 WMMAs per k-step against one staged tile.
__global__ __launch_bounds__(256) void proj_kernel(
    const bf16* __restrict__ qin, const bf16* __restrict__ kin,
    const bf16* __restrict__ vin,
    const bf16* __restrict__ wqb, const bf16* __restrict__ wkb,
    const bf16* __restrict__ wvb,
    const float* __restrict__ bq, const float* __restrict__ bk,
    const float* __restrict__ bv,
    bf16* __restrict__ qbf, bf16* __restrict__ kbf, bf16* __restrict__ vtbf) {
  __shared__ bf16 s_w[2][64][32];

  const int which = blockIdx.z;
  const bf16* in    = (which == 0) ? qin : (which == 1) ? kin : vin;
  const bf16* w     = (which == 0) ? wqb : (which == 1) ? wkb : wvb;
  const float* bias = (which == 0) ? bq  : (which == 1) ? bk  : bv;
  bf16* out         = (which == 0) ? qbf : (which == 1) ? kbf : vtbf;

  const int lane = threadIdx.x & 31;
  const int wv   = threadIdx.x >> 5;
  const int n0b  = ((int)blockIdx.x & 15) * 64;             // 16 N-groups
  const int m0   = (((int)blockIdx.x >> 4) * 8 + wv) * 32;  // 16 M-blocks x 8 waves
  const int c = lane & 15, hh = lane >> 4;

  v8f acc[2][4];
#pragma unroll
  for (int t = 0; t < 4; ++t) {
    const float bv_ = bias[n0b + t * 16 + c];
#pragma unroll
    for (int e = 0; e < 8; ++e) { acc[0][t][e] = bv_; acc[1][t][e] = bv_; }
  }

  stage_issue(w + (size_t)n0b * D_, &s_w[0][0][0], 32, 64, D_, wv);

  for (int ks = 0; ks < D_ / 32; ++ks) {
    const int k0 = ks * 32;
    const int buf = ks & 1;
    stage_wait(wv);
    __syncthreads();
    if (k0 + 32 < D_)
      stage_issue(w + (size_t)n0b * D_ + (k0 + 32), &s_w[buf ^ 1][0][0],
                  32, 64, D_, wv);
    // preload all fragments, then issue the WMMA burst
    v16bf a0 = load_a_bf16(in, D_, m0, k0, lane);
    v16bf a1 = load_a_bf16(in, D_, m0 + 16, k0, lane);
    v16bf b0 = load_b_lds(&s_w[buf][0][0], 32, 0,  0, lane);
    v16bf b1 = load_b_lds(&s_w[buf][0][0], 32, 16, 0, lane);
    v16bf b2 = load_b_lds(&s_w[buf][0][0], 32, 32, 0, lane);
    v16bf b3 = load_b_lds(&s_w[buf][0][0], 32, 48, 0, lane);
    acc[0][0] = wmma_bf16(a0, b0, acc[0][0]);
    acc[1][0] = wmma_bf16(a1, b0, acc[1][0]);
    acc[0][1] = wmma_bf16(a0, b1, acc[0][1]);
    acc[1][1] = wmma_bf16(a1, b1, acc[1][1]);
    acc[0][2] = wmma_bf16(a0, b2, acc[0][2]);
    acc[1][2] = wmma_bf16(a1, b2, acc[1][2]);
    acc[0][3] = wmma_bf16(a0, b3, acc[0][3]);
    acc[1][3] = wmma_bf16(a1, b3, acc[1][3]);
    __syncthreads();
  }

#pragma unroll
  for (int mt = 0; mt < 2; ++mt) {
#pragma unroll
    for (int t = 0; t < 4; ++t) {
      const int n  = n0b + t * 16 + c;
      const int hd = n >> 6;
      const int dk = n & 63;
#pragma unroll
      for (int v = 0; v < 8; ++v) {
        const int m  = m0 + mt * 16 + v + 8 * hh;
        const int bb = m >> 11;
        const int s  = m & (S_ - 1);
        size_t idx;
        if (which == 2) idx = (((size_t)bb * H_ + hd) * DK_ + dk) * S_ + s;  // V^T
        else            idx = (((size_t)bb * H_ + hd) * S_ + s) * DK_ + dk;
        out[idx] = (bf16)acc[mt][t][v];
      }
    }
  }
}

// ---- kernel 2: flash attention; 8 waves share one (b,h)'s K/V tiles --------
__global__ __launch_bounds__(256) void attn_kernel(
    const bf16* __restrict__ qbf, const bf16* __restrict__ kbf,
    const bf16* __restrict__ vtbf, const int* __restrict__ mask,
    bf16* __restrict__ xbf) {
  __shared__ bf16  s_k[2][32][64];   // key tile   [kpos][d]
  __shared__ bf16  s_v[2][64][32];   // value tile [dk][kpos]
  __shared__ float s_sc[8][16][32];
  __shared__ bf16  s_p[8][16][32];
  __shared__ float s_al[8][16];

  const int lane = threadIdx.x & 31;
  const int wv   = threadIdx.x >> 5;
  const int bh   = (int)blockIdx.x >> 4;   // 32 (b,h) pairs
  const int qblk = (int)blockIdx.x & 15;   // 16 q-blocks of 128
  const int hd = bh & 15, bb = bh >> 4;
  const int q0 = qblk * 128 + wv * 16;
  const int c = lane & 15, hh = lane >> 4;

  const bf16* Qp = qbf  + ((size_t)bb * H_ + hd) * S_ * DK_;
  const bf16* Kp = kbf  + ((size_t)bb * H_ + hd) * S_ * DK_;
  const bf16* Vt = vtbf + ((size_t)bb * H_ + hd) * DK_ * S_;
  const int*  mk = mask + (size_t)bb * S_;

  v16bf qa0 = load_a_bf16(Qp, DK_, q0, 0, lane);
  v16bf qa1 = load_a_bf16(Qp, DK_, q0, 32, lane);

  v8f oc[4];
#pragma unroll
  for (int t = 0; t < 4; ++t)
#pragma unroll
    for (int e = 0; e < 8; ++e) oc[t][e] = 0.f;

  float m_r = -__builtin_inff();
  float l_r = 0.f;

  stage_issue(Kp, &s_k[0][0][0], 64, 32, DK_, wv);
  stage_issue(Vt, &s_v[0][0][0], 32, 64, S_, wv);

  for (int ks = 0; ks < S_ / 32; ++ks) {
    const int k0 = ks * 32;
    const int buf = ks & 1;
    stage_wait(wv);
    __syncthreads();
    if (k0 + 32 < S_) {
      stage_issue(Kp + (size_t)(k0 + 32) * DK_, &s_k[buf ^ 1][0][0], 64, 32, DK_, wv);
      stage_issue(Vt + (k0 + 32),               &s_v[buf ^ 1][0][0], 32, 64, S_, wv);
    }

    // scores 16x32 = Q(16x64) . K^T : preload 4 K fragments, burst 4 WMMAs
    v16bf kb00 = load_b_lds(&s_k[buf][0][0], 64, 0,  0,  lane);
    v16bf kb01 = load_b_lds(&s_k[buf][0][0], 64, 0,  32, lane);
    v16bf kb10 = load_b_lds(&s_k[buf][0][0], 64, 16, 0,  lane);
    v16bf kb11 = load_b_lds(&s_k[buf][0][0], 64, 16, 32, lane);
    v8f sc0, sc1;
#pragma unroll
    for (int e = 0; e < 8; ++e) { sc0[e] = 0.f; sc1[e] = 0.f; }
    sc0 = wmma_bf16(qa0, kb00, sc0);
    sc1 = wmma_bf16(qa0, kb10, sc1);
    sc0 = wmma_bf16(qa1, kb01, sc0);
    sc1 = wmma_bf16(qa1, kb11, sc1);

    const bool dead0 = (mk[k0 + c] == 0);
    const bool dead1 = (mk[k0 + 16 + c] == 0);
#pragma unroll
    for (int v = 0; v < 8; ++v) {
      s_sc[wv][v + 8 * hh][c]      = dead0 ? NEGV : sc0[v] * 0.125f;
      s_sc[wv][v + 8 * hh][16 + c] = dead1 ? NEGV : sc1[v] * 0.125f;
    }
    __syncthreads();

    // online softmax: lane r < 16 owns query row r
    if (lane < 16) {
      const int r = lane;
      float tmax = -__builtin_inff();
#pragma unroll 4
      for (int j = 0; j < 32; ++j) tmax = fmaxf(tmax, s_sc[wv][r][j]);
      const float newm  = fmaxf(m_r, tmax);
      const float alpha = __expf(m_r - newm);
      float psum = 0.f;
#pragma unroll 4
      for (int j = 0; j < 32; ++j) {
        const float p = __expf(s_sc[wv][r][j] - newm);
        s_p[wv][r][j] = (bf16)p;
        psum += p;
      }
      l_r = l_r * alpha + psum;
      m_r = newm;
      s_al[wv][r] = alpha;
    }
    __syncthreads();

#pragma unroll
    for (int v = 0; v < 8; ++v) {
      const float al = s_al[wv][v + 8 * hh];
#pragma unroll
      for (int t = 0; t < 4; ++t) oc[t][v] *= al;
    }

    v16bf pa;  // P in A-fragment layout from LDS
#pragma unroll
    for (int v = 0; v < 8; ++v) {
      const int kk = ((v < 4) ? (2 * v) : (8 + 2 * v)) + hh * 8;
      pa[2 * v]     = s_p[wv][c][kk];
      pa[2 * v + 1] = s_p[wv][c][kk + 1];
    }

    // O += P x V : preload 4 V fragments, burst 4 WMMAs
    v16bf vb0 = load_b_lds(&s_v[buf][0][0], 32, 0,  0, lane);
    v16bf vb1 = load_b_lds(&s_v[buf][0][0], 32, 16, 0, lane);
    v16bf vb2 = load_b_lds(&s_v[buf][0][0], 32, 32, 0, lane);
    v16bf vb3 = load_b_lds(&s_v[buf][0][0], 32, 48, 0, lane);
    oc[0] = wmma_bf16(pa, vb0, oc[0]);
    oc[1] = wmma_bf16(pa, vb1, oc[1]);
    oc[2] = wmma_bf16(pa, vb2, oc[2]);
    oc[3] = wmma_bf16(pa, vb3, oc[3]);
    __syncthreads();
  }

  if (lane < 16) s_al[wv][lane] = 1.f / l_r;
  __syncthreads();
#pragma unroll
  for (int v = 0; v < 8; ++v) {
    const int r = v + 8 * hh;
    const float inv = s_al[wv][r];
    const int q = q0 + r;
#pragma unroll
    for (int t = 0; t < 4; ++t) {
      const int col = hd * DK_ + t * 16 + c;
      xbf[((size_t)bb * S_ + q) * D_ + col] = (bf16)(oc[t][v] * inv);
    }
  }
}

// ---- kernel 3: output projection (fp32 result), TDM-staged wo --------------
__global__ __launch_bounds__(256) void out_proj_kernel(
    const bf16* __restrict__ xbf, const bf16* __restrict__ wob,
    const float* __restrict__ bo, float* __restrict__ out) {
  __shared__ bf16 s_w[2][64][32];

  const int lane = threadIdx.x & 31;
  const int wv   = threadIdx.x >> 5;
  const int n0b  = ((int)blockIdx.x & 15) * 64;
  const int m0   = (((int)blockIdx.x >> 4) * 8 + wv) * 32;
  const int c = lane & 15, hh = lane >> 4;

  v8f acc[2][4];
#pragma unroll
  for (int t = 0; t < 4; ++t) {
    const float bv_ = bo[n0b + t * 16 + c];
#pragma unroll
    for (int e = 0; e < 8; ++e) { acc[0][t][e] = bv_; acc[1][t][e] = bv_; }
  }

  stage_issue(wob + (size_t)n0b * D_, &s_w[0][0][0], 32, 64, D_, wv);

  for (int ks = 0; ks < D_ / 32; ++ks) {
    const int k0 = ks * 32;
    const int buf = ks & 1;
    stage_wait(wv);
    __syncthreads();
    if (k0 + 32 < D_)
      stage_issue(wob + (size_t)n0b * D_ + (k0 + 32), &s_w[buf ^ 1][0][0],
                  32, 64, D_, wv);
    v16bf a0 = load_a_bf16(xbf, D_, m0, k0, lane);
    v16bf a1 = load_a_bf16(xbf, D_, m0 + 16, k0, lane);
    v16bf b0 = load_b_lds(&s_w[buf][0][0], 32, 0,  0, lane);
    v16bf b1 = load_b_lds(&s_w[buf][0][0], 32, 16, 0, lane);
    v16bf b2 = load_b_lds(&s_w[buf][0][0], 32, 32, 0, lane);
    v16bf b3 = load_b_lds(&s_w[buf][0][0], 32, 48, 0, lane);
    acc[0][0] = wmma_bf16(a0, b0, acc[0][0]);
    acc[1][0] = wmma_bf16(a1, b0, acc[1][0]);
    acc[0][1] = wmma_bf16(a0, b1, acc[0][1]);
    acc[1][1] = wmma_bf16(a1, b1, acc[1][1]);
    acc[0][2] = wmma_bf16(a0, b2, acc[0][2]);
    acc[1][2] = wmma_bf16(a1, b2, acc[1][2]);
    acc[0][3] = wmma_bf16(a0, b3, acc[0][3]);
    acc[1][3] = wmma_bf16(a1, b3, acc[1][3]);
    __syncthreads();
  }

#pragma unroll
  for (int mt = 0; mt < 2; ++mt)
#pragma unroll
    for (int t = 0; t < 4; ++t)
#pragma unroll
      for (int v = 0; v < 8; ++v)
        out[(size_t)(m0 + mt * 16 + v + 8 * hh) * D_ + n0b + t * 16 + c] =
            acc[mt][t][v];
}

extern "C" void kernel_launch(void* const* d_in, const int* in_sizes, int n_in,
                              void* d_out, int out_size, void* d_ws, size_t ws_size,
                              hipStream_t stream) {
  (void)in_sizes; (void)n_in; (void)out_size; (void)ws_size;
  const float* query = (const float*)d_in[0];
  const float* key   = (const float*)d_in[1];
  const float* value = (const float*)d_in[2];
  const int*   mask  = (const int*)d_in[3];
  const float* wq = (const float*)d_in[4];
  const float* bq = (const float*)d_in[5];
  const float* wk = (const float*)d_in[6];
  const float* bk = (const float*)d_in[7];
  const float* wv = (const float*)d_in[8];
  const float* bv = (const float*)d_in[9];
  const float* wo = (const float*)d_in[10];
  const float* bo = (const float*)d_in[11];

  const size_t NTOK = (size_t)B_ * S_ * D_;  // 4M elements
  const size_t NW   = (size_t)D_ * D_;       // 1M elements
  bf16* p = (bf16*)d_ws;
  bf16* qinb = p; p += NTOK;
  bf16* kinb = p; p += NTOK;
  bf16* vinb = p; p += NTOK;
  bf16* wqb  = p; p += NW;
  bf16* wkb  = p; p += NW;
  bf16* wvb  = p; p += NW;
  bf16* wob  = p; p += NW;
  bf16* qbf  = p; p += NTOK;
  bf16* kbf  = p; p += NTOK;
  bf16* vtbf = p; p += NTOK;
  bf16* xbf  = p; p += NTOK;

  const int actBlocks = (int)(NTOK / 4 / 256);  // 4096
  const int wBlocks   = (int)(NW / 4 / 256);    // 1024
  cvt_bf16_kernel<<<actBlocks, 256, 0, stream>>>(query, qinb, (int)NTOK);
  cvt_bf16_kernel<<<actBlocks, 256, 0, stream>>>(key,   kinb, (int)NTOK);
  cvt_bf16_kernel<<<actBlocks, 256, 0, stream>>>(value, vinb, (int)NTOK);
  cvt_bf16_kernel<<<wBlocks, 256, 0, stream>>>(wq, wqb, (int)NW);
  cvt_bf16_kernel<<<wBlocks, 256, 0, stream>>>(wk, wkb, (int)NW);
  cvt_bf16_kernel<<<wBlocks, 256, 0, stream>>>(wv, wvb, (int)NW);
  cvt_bf16_kernel<<<wBlocks, 256, 0, stream>>>(wo, wob, (int)NW);

  proj_kernel<<<dim3(256, 1, 3), 256, 0, stream>>>(qinb, kinb, vinb,
                                                   wqb, wkb, wvb, bq, bk, bv,
                                                   qbf, kbf, vtbf);
  attn_kernel<<<dim3(512), 256, 0, stream>>>(qbf, kbf, vtbf, mask, xbf);
  out_proj_kernel<<<dim3(256), 256, 0, stream>>>(xbf, wob, bo, (float*)d_out);
}
